// Attention_76656576299266
// MI455X (gfx1250) — matrix operand order repaired and checked by reading, not compiled, not run on hardware
//
#include <hip/hip_runtime.h>
#include <hip/hip_bf16.h>
#include <math.h>
#include <stdint.h>

typedef __attribute__((ext_vector_type(8)))  float        v8f;
typedef __attribute__((ext_vector_type(16))) __bf16       v16bf;
typedef __attribute__((ext_vector_type(4)))  unsigned int v4u;
typedef __attribute__((ext_vector_type(4)))  int          v4i;
typedef __attribute__((ext_vector_type(8)))  int          v8i;

#define N_ENC   2048
#define BATCH   32
#define ENCD    1024
#define HID     1024
#define M_ROWS  (N_ENC * BATCH)    // 65536
#define M_TILE  64                 // rows per workgroup
#define N_TILE  128                // h-chunk per pass
#define K_BLK   32                 // K per step (one bf16 WMMA deep)
#define A_STRIDE 1032              // bf16 elems/row (1024+8 pad): conflict-free ds_load_b128
#define W_STRIDE 40                // bf16 elems/row (32+8 pad)
#define W_BUF_ELEMS (N_TILE * W_STRIDE)   // 5120 bf16 = 10240 B per buffer
#define NEG_BIG 2.0e9f

#if __has_builtin(__builtin_amdgcn_tensor_load_to_lds) && __has_builtin(__builtin_amdgcn_s_wait_tensorcnt)
#define HAVE_TDM 1
#else
#define HAVE_TDM 0
#endif

// ---- fp32 -> bf16 round-to-nearest-even -----------------------------------
__device__ __forceinline__ unsigned short f2bf(float f) {
  unsigned u = __float_as_uint(f);
  u += 0x7FFFu + ((u >> 16) & 1u);
  return (unsigned short)(u >> 16);
}
__device__ __forceinline__ unsigned pack2(float a, float b) {
  return (unsigned)f2bf(a) | ((unsigned)f2bf(b) << 16);
}

// branch-free tanh: sign(x) * (1-e)/(1+e), e = exp(-2|x|)  (one v_exp_f32)
__device__ __forceinline__ float fast_tanh(float x) {
  float ax = fabsf(x);
  float e  = __expf(-2.0f * ax);
  float t  = (1.0f - e) / (1.0f + e);
  return copysignf(t, x);
}

union FragBF { v16bf v; float4 f4[2]; };

// ---------------------------------------------------------------------------
// Kernel 0: w_enc fp32 -> bf16 (row-major [1024,1024]), L2-resident source for TDM
// ---------------------------------------------------------------------------
__global__ __launch_bounds__(256)
void wenc_bf16_kernel(const float* __restrict__ w_enc, __bf16* __restrict__ wbf) {
  int i = blockIdx.x * 256 + threadIdx.x;     // 0 .. 262143 (x4 floats)
  float4 f = *(const float4*)(w_enc + (size_t)i * 4);
  *(uint2*)(wbf + (size_t)i * 4) = make_uint2(pack2(f.x, f.y), pack2(f.z, f.w));
}

// ---------------------------------------------------------------------------
// Kernel 1: proj_dec[b][h] = dec[b,:] . w_dec[h,:]   (67 MFLOP, noise)
// ---------------------------------------------------------------------------
__global__ __launch_bounds__(256)
void proj_dec_kernel(const float* __restrict__ dec,
                     const float* __restrict__ w_dec,
                     float* __restrict__ pd) {
  int idx = blockIdx.x * 256 + threadIdx.x;   // 0 .. 32767
  int b = idx >> 10, h = idx & 1023;
  const float4* dv = (const float4*)(dec + (size_t)b * ENCD);
  const float4* wv = (const float4*)(w_dec + (size_t)h * ENCD);
  float s = 0.f;
#pragma unroll 4
  for (int k = 0; k < ENCD / 4; ++k) {
    float4 a = dv[k], w = wv[k];
    s += a.x * w.x + a.y * w.y + a.z * w.z + a.w * w.w;
  }
  pd[idx] = s;
}

// ---------------------------------------------------------------------------
// TDM descriptor: load 128(h) x 32(k) bf16 tile of wbf into LDS with 16B row pad
// ---------------------------------------------------------------------------
#if HAVE_TDM
__device__ __forceinline__ void tdm_load_w(const __bf16* wbf, unsigned ldsOff,
                                           int hBase, int kBase) {
  unsigned long long ga = (unsigned long long)(uintptr_t)wbf +
                          ((unsigned long long)hBase * ENCD + (unsigned)kBase) * 2ull;
  v4u g0;
  g0.x = 1u;                                     // count=1 (valid user descriptor)
  g0.y = ldsOff;                                 // lds_addr
  g0.z = (unsigned)ga;                           // global_addr[31:0]
  g0.w = ((unsigned)(ga >> 32) & 0x1FFFFFFu) | (2u << 30);  // addr[56:32] | type=2
  // data_size=2B(code1), pad_enable, pad_interval=16DW(code3), pad_amount=4DW(code3)
  unsigned w0 = (1u << 16) | (1u << 20) | (3u << 22) | (3u << 25);
  v8i g1;
  g1[0] = (int)w0;
  g1[1] = (int)((unsigned)ENCD << 16);           // tensor_dim0 = 1024 (lo16 in hi half)
  g1[2] = (int)((unsigned)HID << 16);            // tensor_dim0 hi=0 | tensor_dim1 lo
  g1[3] = (int)((unsigned)K_BLK << 16);          // tensor_dim1 hi=0 | tile_dim0=32
  g1[4] = N_TILE;                                // tile_dim1=128 | tile_dim2=0
  g1[5] = ENCD;                                  // tensor_dim0_stride = 1024
  g1[6] = 0;
  g1[7] = 0;
  v4i z = (v4i)0;
#if defined(__clang_major__) && __clang_major__ >= 23
  __builtin_amdgcn_tensor_load_to_lds(g0, g1, z, z, (v8i)0, 0);
#else
  __builtin_amdgcn_tensor_load_to_lds(g0, g1, z, z, 0);
#endif
}
#endif

// ---------------------------------------------------------------------------
// Kernel 2: fused logits[r] = w_out . tanh( enc[r,:] @ w_enc^T + pd[r%32,:] )
// 128 threads = 4 waves; wave w owns rows [w*16, w*16+16) x all 128 N cols.
// LDS caps occupancy at 2 waves/SIMD -> tell the scheduler so it can keep
// multiple B fragments in flight (partial s_wait_dscnt instead of 0).
// ---------------------------------------------------------------------------
__global__ __launch_bounds__(128)
__attribute__((amdgpu_waves_per_eu(2)))
void attn_logits_kernel(const float* __restrict__ enc,     // [65536,1024] fp32
                        const __bf16* __restrict__ wbf,    // [1024,1024] bf16
                        const float* __restrict__ pd,      // [32,1024]
                        const float* __restrict__ w_out,   // [1024]
                        float* __restrict__ logits) {      // [65536]
  extern __shared__ char smem[];
  __bf16* As   = (__bf16*)smem;                                  // 64 x A_STRIDE
  __bf16* Ws   = (__bf16*)(smem + M_TILE * A_STRIDE * 2);         // 2 x 128 x W_STRIDE
  float*  wout = (float*)(smem + M_TILE * A_STRIDE * 2 + 2 * W_BUF_ELEMS * 2);

  const int tid    = threadIdx.x;
  const int lane   = tid & 31;
  const int wave   = tid >> 5;          // 0..3 -> M sub-tile
  const int laneHi = lane >> 4;
  const int lane15 = lane & 15;
  const int mBase  = blockIdx.x * M_TILE;

#if HAVE_TDM
  const unsigned wLdsBase = (unsigned)(uintptr_t)(void*)Ws;
  if (wave == 0) tdm_load_w(wbf, wLdsBase, 0, 0);   // prefetch first W tile (overlaps A stage)
#endif

  // ---- stage full-K A tile (fp32 -> bf16) into LDS, once ----
  for (int i = tid; i < M_TILE * (ENCD / 4); i += 128) {
    int row = i >> 8, c4 = i & 255;
    float4 f = *(const float4*)(enc + (size_t)(mBase + row) * ENCD + c4 * 4);
    *(uint2*)&As[row * A_STRIDE + c4 * 4] =
        make_uint2(pack2(f.x, f.y), pack2(f.z, f.w));
  }
  for (int i = tid; i < HID; i += 128) wout[i] = w_out[i];

  const float* pdrow[8];
#pragma unroll
  for (int j = 0; j < 8; ++j) {
    int b = (mBase + wave * 16 + j + laneHi * 8) & (BATCH - 1);
    pdrow[j] = pd + (size_t)b * HID;
  }

  float plog[8];
#pragma unroll
  for (int j = 0; j < 8; ++j) plog[j] = 0.f;

  v8f acc[8];
#pragma unroll
  for (int nt = 0; nt < 8; ++nt) acc[nt] = (v8f)0.0f;

#define LOADB(F, NT)                                                   \
  { const __bf16* bp_ = bp0 + (NT) * 16 * W_STRIDE;                    \
    (F).f4[0] = *(const float4*)bp_;                                   \
    (F).f4[1] = *(const float4*)(bp_ + 16); }
#define WMMA(F, NT)                                                    \
  acc[NT] = __builtin_amdgcn_wmma_f32_16x16x32_bf16(                   \
      false, a.v, false, (F).v, (short)0, acc[NT], false, false);

  const int T = (HID / N_TILE) * (ENCD / K_BLK);    // 8 passes * 32 ksteps = 256
  for (int t = 0; t < T; ++t) {
    const int nPass = t >> 5;
    const int kStep = t & 31;
    __bf16* Wb = Ws + (t & 1) * W_BUF_ELEMS;

#if HAVE_TDM
    if (wave == 0) __builtin_amdgcn_s_wait_tensorcnt(0);  // tile t landed
    __syncthreads();
    if (wave == 0 && t + 1 < T)                            // stream tile t+1 into other buf
      tdm_load_w(wbf, wLdsBase + ((t + 1) & 1) * (W_BUF_ELEMS * 2),
                 ((t + 1) >> 5) * N_TILE, ((t + 1) & 31) * K_BLK);
#else
    // fallback: cooperative bf16 copy (no conversion needed)
    for (int i = tid; i < N_TILE * 4; i += 128) {
      int row = i >> 2, c = (i & 3) * 8;
      *(float4*)&Wb[row * W_STRIDE + c] =
          *(const float4*)(wbf + (size_t)(nPass * N_TILE + row) * ENCD + kStep * K_BLK + c);
    }
    __syncthreads();
#endif

    // ---- 2-deep software pipeline: B loads run ahead of the WMMA chain ----
    FragBF a;
    const __bf16* ap = &As[(wave * 16 + lane15) * A_STRIDE + kStep * K_BLK + laneHi * 8];
    a.f4[0] = *(const float4*)ap;
    a.f4[1] = *(const float4*)(ap + 16);

    const __bf16* bp0 = &Wb[lane15 * W_STRIDE + laneHi * 8];
    FragBF f0, f1, f2, f3, f4, f5, f6, f7;
    LOADB(f0, 0)
    LOADB(f1, 1)
    WMMA(f0, 0)  LOADB(f2, 2)
    WMMA(f1, 1)  LOADB(f3, 3)
    WMMA(f2, 2)  LOADB(f4, 4)
    WMMA(f3, 3)  LOADB(f5, 5)
    WMMA(f4, 4)  LOADB(f6, 6)
    WMMA(f5, 5)  LOADB(f7, 7)
    WMMA(f6, 6)
    WMMA(f7, 7)

#if !HAVE_TDM
    __syncthreads();
#endif

    if (kStep == 31) {   // end of N-pass: consume accumulators into partial logits
#pragma unroll
      for (int nt = 0; nt < 8; ++nt) {
        int h = nPass * N_TILE + nt * 16 + lane15;
        float wo = wout[h];
#pragma unroll
        for (int j = 0; j < 8; ++j) {
          float c = acc[nt][j] + pdrow[j][h];
          plog[j] += fast_tanh(c) * wo;
        }
        acc[nt] = (v8f)0.0f;
      }
    }
  }
#undef LOADB
#undef WMMA

  // wave32 cross-lane reduce (16-lane groups), direct store: no cross-wave pass
#pragma unroll
  for (int j = 0; j < 8; ++j) {
    float v = plog[j];
    for (int off = 1; off < 16; off <<= 1) v += __shfl_xor(v, off, 32);
    if (lane15 == 0)
      logits[mBase + wave * 16 + laneHi * 8 + j] = v;
  }
}

// ---------------------------------------------------------------------------
// Kernel 3: masked softmax over the sequence dim, one block per batch column
// ---------------------------------------------------------------------------
__global__ __launch_bounds__(256)
void softmax_kernel(const float* __restrict__ logits,
                    const int* __restrict__ mask,
                    float* __restrict__ wgt) {              // [2048,32]
  __shared__ float sred[256];
  int b = blockIdx.x, tid = threadIdx.x;
  float ml[N_ENC / 256];
  float mx = -INFINITY;
#pragma unroll
  for (int i = 0; i < N_ENC / 256; ++i) {
    int idx = (tid + i * 256) * BATCH + b;
    float lg = logits[idx];
    ml[i] = mask[idx] ? lg : -NEG_BIG;
    mx = fmaxf(mx, ml[i]);
  }
  sred[tid] = mx; __syncthreads();
  for (int off = 128; off > 0; off >>= 1) {
    if (tid < off) sred[tid] = fmaxf(sred[tid], sred[tid + off]);
    __syncthreads();
  }
  mx = sred[0]; __syncthreads();

  float ex[N_ENC / 256], sum = 0.f;
#pragma unroll
  for (int i = 0; i < N_ENC / 256; ++i) { ex[i] = expf(ml[i] - mx); sum += ex[i]; }
  sred[tid] = sum; __syncthreads();
  for (int off = 128; off > 0; off >>= 1) {
    if (tid < off) sred[tid] += sred[tid + off];
    __syncthreads();
  }
  float inv = 1.0f / sred[0];
#pragma unroll
  for (int i = 0; i < N_ENC / 256; ++i)
    wgt[(tid + i * 256) * BATCH + b] = ex[i] * inv;
}

// ---------------------------------------------------------------------------
// Kernel 4: attn_response[b][e] = sum_s enc[s][b][e] * wgt[s][b]  (mem-bound)
// ---------------------------------------------------------------------------
__global__ __launch_bounds__(256)
void response_kernel(const float* __restrict__ enc,
                     const float* __restrict__ wgt,
                     float* __restrict__ resp) {
  int b = blockIdx.x >> 2;
  int e = (blockIdx.x & 3) * 256 + threadIdx.x;
  float acc = 0.f;
#pragma unroll 4
  for (int s = 0; s < N_ENC; ++s)
    acc += enc[((size_t)s * BATCH + b) * ENCD + e] * wgt[s * BATCH + b];
  resp[(size_t)b * ENCD + e] = acc;
}

// ---------------------------------------------------------------------------
extern "C" void kernel_launch(void* const* d_in, const int* in_sizes, int n_in,
                              void* d_out, int out_size, void* d_ws, size_t ws_size,
                              hipStream_t stream) {
  const float* enc   = (const float*)d_in[0];
  const int*   mask  = (const int*)d_in[1];
  const float* dec   = (const float*)d_in[2];
  const float* w_enc = (const float*)d_in[3];
  const float* w_dec = (const float*)d_in[4];
  const float* w_out = (const float*)d_in[5];
  float* out = (float*)d_out;

  float*  pd     = (float*)d_ws;                 // [32,1024]
  float*  logits = pd + BATCH * HID;             // [65536]
  __bf16* wbf    = (__bf16*)(logits + M_ROWS);   // [1024,1024] bf16
  float*  wgt    = out + BATCH * ENCD;           // attn_weights region of d_out

  wenc_bf16_kernel<<<HID * ENCD / 1024, 256, 0, stream>>>(w_enc, wbf);
  proj_dec_kernel<<<BATCH * HID / 256, 256, 0, stream>>>(dec, w_dec, pd);

  size_t smem = (size_t)M_TILE * A_STRIDE * 2    // A tile (bf16)      132096
              + (size_t)2 * W_BUF_ELEMS * 2      // W ping/pong         20480
              + (size_t)HID * 4;                 // w_out                4096
  attn_logits_kernel<<<M_ROWS / M_TILE, 128, smem, stream>>>(enc, wbf, pd,
                                                             w_out, logits);
  softmax_kernel<<<BATCH, 256, 0, stream>>>(logits, mask, wgt);
  response_kernel<<<BATCH * 4, 256, 0, stream>>>(enc, wgt, out);
}
